// SA_MambaLayer_64802466562245
// MI455X (gfx1250) — matrix operand order repaired and checked by reading, not compiled, not run on hardware
//
#include <hip/hip_runtime.h>
#include <hip/hip_bf16.h>

// ---- problem constants (from reference) ----
#define DIMC     512
#define D_INNER  1024
#define D_STATE  64
#define DT_RANKC 32
#define D_CONVC  4
#define B_SZC    2
#define N_LENC   2048
#define NTOK     (B_SZC * N_LENC)         // 4096 tokens
#define XPROJ_N  (DT_RANKC + 2 * D_STATE) // 160

typedef _Float16 half16 __attribute__((ext_vector_type(16)));
typedef _Float16 half8  __attribute__((ext_vector_type(8)));
typedef float    f32x8  __attribute__((ext_vector_type(8)));

// ---------------------------------------------------------------------------
// LayerNorm over last dim (512) -> f16 activation (WMMA A-operand)
// one block (256 thr) per token row
// ---------------------------------------------------------------------------
__global__ void ln_to_f16_kernel(const float* __restrict__ x,
                                 const float* __restrict__ w,
                                 const float* __restrict__ b,
                                 _Float16* __restrict__ xn) {
  __shared__ float red[256];
  const int row = blockIdx.x;
  const int t = threadIdx.x;
  const float* xr = x + (size_t)row * DIMC;
  float v0 = xr[t], v1 = xr[t + 256];
  red[t] = v0 + v1;
  __syncthreads();
  for (int s = 128; s > 0; s >>= 1) { if (t < s) red[t] += red[t + s]; __syncthreads(); }
  const float mean = red[0] * (1.0f / DIMC);
  __syncthreads();
  const float d0 = v0 - mean, d1 = v1 - mean;
  red[t] = d0 * d0 + d1 * d1;
  __syncthreads();
  for (int s = 128; s > 0; s >>= 1) { if (t < s) red[t] += red[t + s]; __syncthreads(); }
  const float rstd = rsqrtf(red[0] * (1.0f / DIMC) + 1e-5f);
  _Float16* xo = xn + (size_t)row * DIMC;
  xo[t]       = (_Float16)(d0 * rstd * w[t] + b[t]);
  xo[t + 256] = (_Float16)(d1 * rstd * w[t + 256] + b[t + 256]);
}

// ---------------------------------------------------------------------------
// generic f32 -> f16 conversion (weights)
// ---------------------------------------------------------------------------
__global__ void f32_to_f16_kernel(const float* __restrict__ src,
                                  _Float16* __restrict__ dst, int n) {
  int i = blockIdx.x * 256 + threadIdx.x;
  if (i < n) dst[i] = (_Float16)src[i];
}

// ---------------------------------------------------------------------------
// WMMA GEMM: out[M,Nout] = A[M,K](f16) * W[Nout,K](f16)^T, f32 accumulate.
// Each wave computes a 32x32 output tile (2 M-tiles x 2 N-tiles, 4 accums)
// so every A/B fragment feeds two v_wmma ops (2 b128 loads per wmma).
// 4 waves per block stacked in M -> block tile 128(M) x 32(N).
// grid = (Nout/32, M/128). K must be a multiple of 32 (512/1024/32 here).
// Fragments: 16-bit A 16x32 / B 32x16 use contiguous k = 16*lhi + s per lane; f32 C/D per CDNA5 ISA 7.12.2.
// EPI: 0 = plain store, 1 = softplus(x + bias[n]) (dt_proj epilogue)
// ---------------------------------------------------------------------------
template <int EPI>
__global__ void wmma_gemm_kernel(const _Float16* __restrict__ A,
                                 const _Float16* __restrict__ W,
                                 const float* __restrict__ bias,
                                 float* __restrict__ out,
                                 int M, int Nout, int K) {
  const int wave = threadIdx.x >> 5;
  const int lane = threadIdx.x & 31;
  const int lrow = lane & 15;   // A: m in tile | B: n in tile
  const int lhi  = lane >> 4;   // half-wave select
  const int nBase = blockIdx.x * 32;
  const int mBase = (blockIdx.y * 4 + wave) * 32;

  // A lane holds row (mBase+lrow); K chunk {16*lhi..+15} per k-step (contiguous)
  const _Float16* arow0 = A + (size_t)(mBase + lrow) * K + 16 * lhi;
  const _Float16* arow1 = arow0 + (size_t)16 * K;
  // B lane holds column (nBase+lrow); K chunk {16*lhi..+15} per k-step (contiguous)
  const _Float16* brow0 = W + (size_t)(nBase + lrow) * K + 16 * lhi;
  const _Float16* brow1 = brow0 + (size_t)16 * K;

  f32x8 c00 = {}, c01 = {}, c10 = {}, c11 = {};
  for (int k0 = 0; k0 < K; k0 += 32) {
    half8 a0lo = *(const half8*)(arow0 + k0);
    half8 a0hi = *(const half8*)(arow0 + k0 + 8);
    half8 a1lo = *(const half8*)(arow1 + k0);
    half8 a1hi = *(const half8*)(arow1 + k0 + 8);
    half16 a0 = __builtin_shufflevector(a0lo, a0hi, 0, 1, 2, 3, 4, 5, 6, 7,
                                                    8, 9, 10, 11, 12, 13, 14, 15);
    half16 a1 = __builtin_shufflevector(a1lo, a1hi, 0, 1, 2, 3, 4, 5, 6, 7,
                                                    8, 9, 10, 11, 12, 13, 14, 15);
    half16 b0 = *(const half16*)(brow0 + k0);
    half16 b1 = *(const half16*)(brow1 + k0);
    c00 = __builtin_amdgcn_wmma_f32_16x16x32_f16(false, a0, false, b0, (short)0, c00, false, false);
    c01 = __builtin_amdgcn_wmma_f32_16x16x32_f16(false, a0, false, b1, (short)0, c01, false, false);
    c10 = __builtin_amdgcn_wmma_f32_16x16x32_f16(false, a1, false, b0, (short)0, c10, false, false);
    c11 = __builtin_amdgcn_wmma_f32_16x16x32_f16(false, a1, false, b1, (short)0, c11, false, false);
  }

  const int n0 = nBase + lrow;
  const int n1 = n0 + 16;
  const float bv0 = (EPI == 1) ? bias[n0] : 0.0f;
  const float bv1 = (EPI == 1) ? bias[n1] : 0.0f;
#pragma unroll
  for (int j = 0; j < 8; ++j) {
    const int m0 = mBase + j + 8 * lhi;  // C/D layout: VGPR j -> M=j (lanes 0-15) / M=j+8
    const int m1 = m0 + 16;
    float v00 = c00[j], v01 = c01[j], v10 = c10[j], v11 = c11[j];
    if (EPI == 1) {                      // softplus(v + bias)
      v00 += bv0; v00 = (v00 > 20.0f) ? v00 : log1pf(expf(v00));
      v01 += bv1; v01 = (v01 > 20.0f) ? v01 : log1pf(expf(v01));
      v10 += bv0; v10 = (v10 > 20.0f) ? v10 : log1pf(expf(v10));
      v11 += bv1; v11 = (v11 > 20.0f) ? v11 : log1pf(expf(v11));
    }
    out[(size_t)m0 * Nout + n0] = v00;
    out[(size_t)m0 * Nout + n1] = v01;
    out[(size_t)m1 * Nout + n0] = v10;
    out[(size_t)m1 * Nout + n1] = v11;
  }
}

// ---------------------------------------------------------------------------
// causal depthwise conv1d (d_conv=4) + bias + SiLU over xi (= first half of xz)
// one thread per (token,channel) element; emits f32 (scan input) and f16 (WMMA A)
// ---------------------------------------------------------------------------
__global__ void conv_silu_kernel(const float* __restrict__ xz,
                                 const float* __restrict__ cw,
                                 const float* __restrict__ cb,
                                 float* __restrict__ xi32,
                                 _Float16* __restrict__ xi16) {
  const int i = blockIdx.x * 256 + threadIdx.x;  // NTOK * D_INNER threads
  const int row = i >> 10;                       // token index b*N+n
  const int d = i & (D_INNER - 1);
  const int b = row >> 11;                       // N_LEN = 2048
  const int n = row & (N_LENC - 1);
  float acc = cb[d];
#pragma unroll
  for (int j = 0; j < D_CONVC; ++j) {
    const int t = n - (D_CONVC - 1) + j;
    if (t >= 0)
      acc += cw[d * D_CONVC + j] * xz[(size_t)(b * N_LENC + t) * (2 * D_INNER) + d];
  }
  const float s = acc / (1.0f + expf(-acc));     // SiLU
  xi32[i] = s;
  xi16[i] = (_Float16)s;
}

// ---------------------------------------------------------------------------
// extract dt slice (first 32 cols of x_dbl) as f16 A-operand for dt_proj GEMM
// ---------------------------------------------------------------------------
__global__ void dt_extract_kernel(const float* __restrict__ x_dbl,
                                  _Float16* __restrict__ dt16) {
  const int i = blockIdx.x * 256 + threadIdx.x;  // NTOK * 32
  const int row = i >> 5;
  const int r = i & 31;
  dt16[i] = (_Float16)x_dbl[(size_t)row * XPROJ_N + r];
}

// ---------------------------------------------------------------------------
// selective scan: one wave32 per (b, d) channel; lane holds states s and s+32.
// h_s[n] = exp(dt*A_s) h_s[n-1] + dt*u*B_s[n];  y = sum_s h_s * C_s[n]
// Software-pipelined: iteration n+1's operand loads are issued before the
// dependent exp/FMA/reduce chain of iteration n, hiding load latency behind
// the serial recurrence.
// fused epilogue: y = (y + D*u) * SiLU(z), stored as f16 (out_proj A-operand)
// ---------------------------------------------------------------------------
__global__ void scan_kernel(const float* __restrict__ x_dbl,  // (NTOK,160): [32,96)=B, [96,160)=C
                            const float* __restrict__ dtf,    // (NTOK, D_INNER)
                            const float* __restrict__ u,      // (NTOK, D_INNER) post-conv SiLU
                            const float* __restrict__ xz,     // (NTOK, 2*D_INNER): z at +D_INNER
                            const float* __restrict__ A_log,  // (D_INNER, 64)
                            const float* __restrict__ Dp,     // (D_INNER)
                            _Float16* __restrict__ y16) {     // (NTOK, D_INNER)
  const int wave = threadIdx.x >> 5;
  const int lane = threadIdx.x & 31;
  const int ch = blockIdx.x * 8 + wave;          // 0 .. B*D_INNER-1
  const int b = ch >> 10;
  const int d = ch & (D_INNER - 1);

  const float a0 = -expf(A_log[d * D_STATE + lane]);
  const float a1 = -expf(A_log[d * D_STATE + 32 + lane]);
  const float dcoef = Dp[d];
  float h0 = 0.0f, h1 = 0.0f;

  // prologue: load operands for n = 0
  size_t row = (size_t)b * N_LENC;
  float dt_c = dtf[row * D_INNER + d];
  float uu_c = u[row * D_INNER + d];
  const float* xr = x_dbl + row * XPROJ_N;
  float B0_c = xr[32 + lane], B1_c = xr[64 + lane];
  float C0_c = xr[96 + lane], C1_c = xr[128 + lane];
  float z_c = xz[row * (2 * D_INNER) + D_INNER + d];

  for (int n = 0; n < N_LENC; ++n) {
    // issue next iteration's independent loads first (clamped at the end)
    const int np = (n + 1 < N_LENC) ? (n + 1) : n;
    const size_t rowp = (size_t)b * N_LENC + np;
    const float dt_n = dtf[rowp * D_INNER + d];
    const float uu_n = u[rowp * D_INNER + d];
    const float* xrp = x_dbl + rowp * XPROJ_N;
    const float B0_n = xrp[32 + lane], B1_n = xrp[64 + lane];
    const float C0_n = xrp[96 + lane], C1_n = xrp[128 + lane];
    const float z_n = xz[rowp * (2 * D_INNER) + D_INNER + d];

    // recurrence for step n
    const float dBu = dt_c * uu_c;
    h0 = expf(dt_c * a0) * h0 + dBu * B0_c;
    h1 = expf(dt_c * a1) * h1 + dBu * B1_c;
    float p = h0 * C0_c + h1 * C1_c;
#pragma unroll
    for (int off = 16; off >= 1; off >>= 1) p += __shfl_xor(p, off, 32);
    if (lane == 0) {
      const size_t rw = (size_t)b * N_LENC + n;
      const float yg = (p + uu_c * dcoef) * (z_c / (1.0f + expf(-z_c)));
      y16[rw * D_INNER + d] = (_Float16)yg;
    }

    dt_c = dt_n; uu_c = uu_n;
    B0_c = B0_n; B1_c = B1_n; C0_c = C0_n; C1_c = C1_n;
    z_c = z_n;
  }
}

// ---------------------------------------------------------------------------
extern "C" void kernel_launch(void* const* d_in, const int* in_sizes, int n_in,
                              void* d_out, int out_size, void* d_ws, size_t ws_size,
                              hipStream_t stream) {
  const float* x          = (const float*)d_in[0];
  const float* norm_w     = (const float*)d_in[1];
  const float* norm_b     = (const float*)d_in[2];
  const float* in_proj_w  = (const float*)d_in[3];
  const float* conv_w     = (const float*)d_in[4];
  const float* conv_b     = (const float*)d_in[5];
  const float* x_proj_w   = (const float*)d_in[6];
  const float* dt_proj_w  = (const float*)d_in[7];
  const float* dt_proj_b  = (const float*)d_in[8];
  const float* A_log      = (const float*)d_in[9];
  const float* Dp         = (const float*)d_in[10];
  const float* out_proj_w = (const float*)d_in[11];
  (void)in_sizes; (void)n_in; (void)out_size; (void)ws_size;

  char* ws = (char*)d_ws;
  size_t off = 0;
  auto alloc = [&](size_t bytes) -> char* {
    char* p = ws + off;
    off += (bytes + 255) & ~(size_t)255;
    return p;
  };

  _Float16* xn16    = (_Float16*)alloc((size_t)NTOK * DIMC * 2);
  _Float16* w_in16  = (_Float16*)alloc((size_t)2 * D_INNER * DIMC * 2);
  _Float16* w_xp16  = (_Float16*)alloc((size_t)XPROJ_N * D_INNER * 2);
  _Float16* w_dt16  = (_Float16*)alloc((size_t)D_INNER * DT_RANKC * 2);
  _Float16* w_out16 = (_Float16*)alloc((size_t)DIMC * D_INNER * 2);
  float*    xz      = (float*)alloc((size_t)NTOK * 2 * D_INNER * 4);
  float*    xi32    = (float*)alloc((size_t)NTOK * D_INNER * 4);
  _Float16* xi16    = (_Float16*)alloc((size_t)NTOK * D_INNER * 2);
  float*    x_dbl   = (float*)alloc((size_t)NTOK * XPROJ_N * 4);
  _Float16* dt16    = (_Float16*)alloc((size_t)NTOK * DT_RANKC * 2);
  float*    dt_full = (float*)alloc((size_t)NTOK * D_INNER * 4);
  _Float16* y16     = (_Float16*)alloc((size_t)NTOK * D_INNER * 2);

  // 1) LayerNorm -> f16
  ln_to_f16_kernel<<<NTOK, 256, 0, stream>>>(x, norm_w, norm_b, xn16);

  // 2) weight conversions to f16
  f32_to_f16_kernel<<<(2 * D_INNER * DIMC + 255) / 256, 256, 0, stream>>>(in_proj_w, w_in16, 2 * D_INNER * DIMC);
  f32_to_f16_kernel<<<(XPROJ_N * D_INNER + 255) / 256, 256, 0, stream>>>(x_proj_w, w_xp16, XPROJ_N * D_INNER);
  f32_to_f16_kernel<<<(D_INNER * DT_RANKC + 255) / 256, 256, 0, stream>>>(dt_proj_w, w_dt16, D_INNER * DT_RANKC);
  f32_to_f16_kernel<<<(DIMC * D_INNER + 255) / 256, 256, 0, stream>>>(out_proj_w, w_out16, DIMC * D_INNER);

  // 3) in_proj: (4096 x 512) * (2048 x 512)^T -> xz (4096 x 2048)
  wmma_gemm_kernel<0><<<dim3(2 * D_INNER / 32, NTOK / 128), 128, 0, stream>>>(
      xn16, w_in16, nullptr, xz, NTOK, 2 * D_INNER, DIMC);

  // 4) causal depthwise conv + SiLU
  conv_silu_kernel<<<(NTOK * D_INNER) / 256, 256, 0, stream>>>(xz, conv_w, conv_b, xi32, xi16);

  // 5) x_proj: (4096 x 1024) * (160 x 1024)^T -> x_dbl (4096 x 160)
  wmma_gemm_kernel<0><<<dim3(XPROJ_N / 32, NTOK / 128), 128, 0, stream>>>(
      xi16, w_xp16, nullptr, x_dbl, NTOK, XPROJ_N, D_INNER);

  // 6) dt slice -> f16
  dt_extract_kernel<<<(NTOK * DT_RANKC) / 256, 256, 0, stream>>>(x_dbl, dt16);

  // 7) dt_proj + softplus: (4096 x 32) * (1024 x 32)^T -> dt_full (4096 x 1024)
  wmma_gemm_kernel<1><<<dim3(D_INNER / 32, NTOK / 128), 128, 0, stream>>>(
      dt16, w_dt16, dt_proj_b, dt_full, NTOK, D_INNER, DT_RANKC);

  // 8) selective scan + gating -> y16
  scan_kernel<<<(B_SZC * D_INNER) / 8, 256, 0, stream>>>(
      x_dbl, dt_full, xi32, xz, A_log, Dp, y16);

  // 9) out_proj: (4096 x 1024) * (512 x 1024)^T -> out (4096 x 512)
  wmma_gemm_kernel<0><<<dim3(DIMC / 32, NTOK / 128), 128, 0, stream>>>(
      y16, w_out16, nullptr, (float*)d_out, NTOK, DIMC, D_INNER);
}
